// ViT_26087631356349
// MI455X (gfx1250) — matrix operand order repaired and checked by reading, not compiled, not run on hardware
//
#include <hip/hip_runtime.h>
#include <hip/hip_bf16.h>
#include <math.h>

// ---------------- constants ----------------
constexpr int BATCH = 16;
constexpr int CIN   = 3;
constexpr int IMGH  = 224;
constexpr int IMGW  = 224;
constexpr int PATCH = 16;
constexpr int D     = 768;
constexpr int DFF   = 3072;
constexpr int NH    = 12;
constexpr int HD    = 64;
constexpr int LAYERS= 12;
constexpr int NCLS  = 1000;
constexpr int S     = 197;          // 14*14 + 1
constexpr int NP    = 196;
constexpr int MROWS = BATCH * S;    // 3152
constexpr int PSTR  = 224;          // padded P row stride (halfs), 7*32, 16B aligned

typedef _Float16 v16h_t __attribute__((ext_vector_type(16)));
typedef _Float16 v8h_t  __attribute__((ext_vector_type(8)));
typedef float    v8f_t  __attribute__((ext_vector_type(8)));
typedef int      v4i_gcc __attribute__((vector_size(16)));   // matches builtin param type

// ---- CDNA5 async Global->LDS path (guarded; fallback = load + ds_store) ----
#if defined(__has_builtin)
#  if __has_builtin(__builtin_amdgcn_global_load_async_to_lds_b128)
#    define HAS_ASYNC_LDS 1
#  else
#    define HAS_ASYNC_LDS 0
#  endif
#else
#  define HAS_ASYNC_LDS 0
#endif

__device__ inline void cp16_g2l(const _Float16* g, _Float16* l) {
#if HAS_ASYNC_LDS
  __builtin_amdgcn_global_load_async_to_lds_b128((v4i_gcc*)g, (v4i_gcc*)l, 0, 0);
#else
  *(v8h_t*)l = *(const v8h_t*)g;
#endif
}
__device__ inline void async_wait0() {
#if HAS_ASYNC_LDS
#  if __has_builtin(__builtin_amdgcn_s_wait_asynccnt)
  __builtin_amdgcn_s_wait_asynccnt(0);
#  else
  asm volatile("s_wait_asynccnt 0" ::: "memory");
#  endif
#endif
}

// ---- WMMA fragment builders (CDNA5 ISA 7.12.2, 16x16x32 f16) ----
__device__ inline v16h_t make_a_frag(const _Float16* __restrict__ Arow, int kb) {
  v8h_t lo = *(const v8h_t*)(Arow + 2 * kb);        // K = 2kb..2kb+7
  v8h_t hi = *(const v8h_t*)(Arow + 2 * kb + 8);    // K = 2kb+8..2kb+15
  v16h_t a;
#pragma unroll
  for (int i = 0; i < 8; ++i) { a[i] = lo[i]; a[8 + i] = hi[i]; }
  return a;
}
__device__ inline v16h_t make_b_frag(const _Float16* __restrict__ Wrow, int kb2) {
  v8h_t lo = *(const v8h_t*)(Wrow + kb2);       // K = kb2..kb2+7
  v8h_t hi = *(const v8h_t*)(Wrow + kb2 + 8);   // K = kb2+8..kb2+15
  v16h_t b;
#pragma unroll
  for (int i = 0; i < 8; ++i) { b[i] = lo[i]; b[8 + i] = hi[i]; }
  return b;
}

// ---------------- elementwise helpers ----------------
__global__ void k_cvt_f32_f16(const float* __restrict__ src, _Float16* __restrict__ dst, int n) {
  int i = blockIdx.x * 256 + threadIdx.x;
  if (i < n) dst[i] = (_Float16)src[i];
}

__global__ void k_im2col(const float* __restrict__ x, _Float16* __restrict__ A) {
  int idx = blockIdx.x * 256 + threadIdx.x;
  if (idx >= BATCH * NP * D) return;
  int d = idx % D;  int r = idx / D;
  int b = r / NP;   int p = r % NP;
  int py = p / 14, px = p % 14;
  int c = d >> 8;  int rem = d & 255;  int ii = rem >> 4;  int jj = rem & 15;
  float v = x[(((size_t)b * CIN + c) * IMGH + py * PATCH + ii) * IMGW + px * PATCH + jj];
  A[idx] = (_Float16)v;
}

__global__ void k_build_h(const float* __restrict__ patch, const float* __restrict__ cls,
                          const float* __restrict__ pos, float* __restrict__ h) {
  int idx = blockIdx.x * 256 + threadIdx.x;
  if (idx >= MROWS * D) return;
  int d = idx % D;  int row = idx / D;
  int b = row / S;  int s = row % S;
  float v = (s == 0) ? cls[d] : patch[((size_t)b * NP + (s - 1)) * D + d];
  h[idx] = v + pos[(size_t)s * D + d];
}

// LayerNorm over D=768, output f16. Input row r maps to x row r*inStride.
__global__ __launch_bounds__(256)
void k_layernorm_f16(const float* __restrict__ x, const float* __restrict__ g,
                     const float* __restrict__ bta, _Float16* __restrict__ out, int inStride) {
  __shared__ float s1[256], s2[256];
  int r = blockIdx.x;  int t = threadIdx.x;
  const float* xr = x + (size_t)r * inStride * D;
  float lsum = 0.f, lsq = 0.f;
  for (int d = t; d < D; d += 256) { float v = xr[d]; lsum += v; lsq += v * v; }
  s1[t] = lsum; s2[t] = lsq; __syncthreads();
  for (int off = 128; off > 0; off >>= 1) {
    if (t < off) { s1[t] += s1[t + off]; s2[t] += s2[t + off]; }
    __syncthreads();
  }
  float mu = s1[0] / (float)D;
  float var = s2[0] / (float)D - mu * mu;
  float rstd = rsqrtf(var + 1e-6f);
  for (int d = t; d < D; d += 256)
    out[(size_t)r * D + d] = (_Float16)((xr[d] - mu) * rstd * g[d] + bta[d]);
}

// ---------------- WMMA GEMM with async-LDS double buffering ----------------
// C[M,N] = A[M,K] * W[N,K]^T + bias.
// Block: 256 thr = 8 waves (4 M x 2 N); block tile 64x128; wave tile 16x64.
// LDS: A 2x64x40 halfs, B 2x128x40 halfs (row stride 40 -> conflict-free frag reads).
// mode: 0=store f16, 1=store f32, 2=f32 +=, 3=GELU(exact)->f16
__global__ __launch_bounds__(256)
void k_gemm_wmma(const _Float16* __restrict__ A, const _Float16* __restrict__ Wt,
                 const float* __restrict__ bias, float* __restrict__ outF,
                 _Float16* __restrict__ outH, int M, int N, int K, int mode) {
  __shared__ _Float16 sA[2][64][40];
  __shared__ _Float16 sB[2][128][40];

  int tid  = threadIdx.x;
  int lane = tid & 31;
  int wid  = tid >> 5;
  int wr = wid & 3, wc = wid >> 2;
  int lm = lane & 15;
  int M0blk = blockIdx.y * 64, N0blk = blockIdx.x * 128;
  int M0 = M0blk + wr * 16, N0 = N0blk + wc * 64;

  // cooperative stage task mapping (per thread): A: 1 chunk, B: 2 chunks of 8 halfs
  int arow = tid >> 2, ach = (tid & 3) * 8;
  int am = M0blk + arow; if (am >= M) am = M - 1;
  const _Float16* agp = A + (size_t)am * K + ach;
  int brow0 = tid >> 2,           bch0 = (tid & 3) * 8;
  int brow1 = (tid + 256) >> 2,   bch1 = bch0;
  int bn0 = N0blk + brow0; if (bn0 >= N) bn0 = N - 1;
  int bn1 = N0blk + brow1; if (bn1 >= N) bn1 = N - 1;
  const _Float16* bgp0 = Wt + (size_t)bn0 * K + bch0;
  const _Float16* bgp1 = Wt + (size_t)bn1 * K + bch1;

  auto stage = [&](int buf, int kk) {
    cp16_g2l(agp + kk, &sA[buf][arow][ach]);
    cp16_g2l(bgp0 + kk, &sB[buf][brow0][bch0]);
    cp16_g2l(bgp1 + kk, &sB[buf][brow1][bch1]);
  };

  v8f_t acc[4] = {};
  int kb  = (lane >= 16) ? 8 : 0;
  int kb2 = (lane >= 16) ? 16 : 0;

  int buf = 0;
  stage(0, 0);
  async_wait0();
  __syncthreads();

  for (int kk = 0; kk < K; kk += 32) {
    bool more = (kk + 32) < K;
    int nbuf = buf ^ 1;
    if (more) stage(nbuf, kk + 32);   // async fill of next buffer overlaps WMMA

    v16h_t a = make_a_frag(&sA[buf][wr * 16 + lm][0], kb);
#pragma unroll
    for (int j = 0; j < 4; ++j) {
      v16h_t b = make_b_frag(&sB[buf][wc * 64 + j * 16 + lm][0], kb2);
      acc[j] = __builtin_amdgcn_wmma_f32_16x16x32_f16(false, a, false, b,
                                                      (short)0, acc[j], false, false);
    }
    if (more) async_wait0();
    __syncthreads();
    buf = nbuf;
  }

  int mbase = M0 + ((lane >> 4) << 3);
#pragma unroll
  for (int j = 0; j < 4; ++j) {
    int n = N0 + j * 16 + lm;
    if (n >= N) continue;
    float bv = bias ? bias[n] : 0.f;
#pragma unroll
    for (int r = 0; r < 8; ++r) {
      int mm = mbase + r;
      if (mm >= M) continue;
      float v = acc[j][r] + bv;
      size_t idx = (size_t)mm * N + n;
      if      (mode == 0) outH[idx] = (_Float16)v;
      else if (mode == 1) outF[idx] = v;
      else if (mode == 2) outF[idx] += v;
      else { float ge = 0.5f * v * (1.f + erff(v * 0.70710678118f)); outH[idx] = (_Float16)ge; }
    }
  }
}

// ---------------- attention: scores = softmax(Q K^T / 8) ----------------
// grid (B*NH, 13 m-tiles), block 128 (4 waves). q/k layout [B, S, NH, HD].
// Writes f32 scores (stride S) to d_out and f16 P (row stride PSTR, pad zero-filled).
__global__ __launch_bounds__(128)
void k_attn_scores(const _Float16* __restrict__ q, const _Float16* __restrict__ kbuf,
                   float* __restrict__ scoresL, _Float16* __restrict__ Pbuf) {
  __shared__ float sc[16][208];
  __shared__ float red[16][8];
  __shared__ float rowmax[16], rowsum[16];
  const float scale = 0.125f;  // 1/sqrt(64)

  int bh = blockIdx.x;  int b = bh / NH;  int hh = bh % NH;
  int M0 = blockIdx.y * 16;
  int lane = threadIdx.x & 31;  int wv = threadIdx.x >> 5;
  int lm = lane & 15;
  int kb  = (lane >= 16) ? 8 : 0;
  int kb2 = (lane >= 16) ? 16 : 0;

  int m = M0 + lm;  int sm = (m < S) ? m : S - 1;
  const _Float16* qrow = q + ((size_t)(b * S + sm) * D + hh * HD);

  for (int nt = wv; nt < 13; nt += 4) {
    int n = nt * 16 + lm;  int sn = (n < S) ? n : S - 1;
    const _Float16* krow = kbuf + ((size_t)(b * S + sn) * D + hh * HD);
    v8f_t acc = {};
#pragma unroll
    for (int kk = 0; kk < HD; kk += 32) {
      v16h_t a  = make_a_frag(qrow + kk, kb);
      v16h_t bf = make_b_frag(krow + kk, kb2);
      acc = __builtin_amdgcn_wmma_f32_16x16x32_f16(false, a, false, bf,
                                                   (short)0, acc, false, false);
    }
    int rbase = (lane >> 4) << 3;
#pragma unroll
    for (int r = 0; r < 8; ++r) sc[rbase + r][nt * 16 + lm] = acc[r] * scale;
  }
  __syncthreads();

  // block softmax over 16 rows x 197 valid cols (8 col-groups per row)
  int rr = threadIdx.x & 15, gg = threadIdx.x >> 4;
  float mx = -3.0e38f;
  for (int n = gg; n < S; n += 8) mx = fmaxf(mx, sc[rr][n]);
  red[rr][gg] = mx; __syncthreads();
  if (gg == 0) { float m2 = red[rr][0]; for (int i = 1; i < 8; ++i) m2 = fmaxf(m2, red[rr][i]); rowmax[rr] = m2; }
  __syncthreads();
  float psum = 0.f;
  for (int n = gg; n < S; n += 8) { float e = expf(sc[rr][n] - rowmax[rr]); sc[rr][n] = e; psum += e; }
  red[rr][gg] = psum; __syncthreads();
  if (gg == 0) { float s2 = 0.f; for (int i = 0; i < 8; ++i) s2 += red[rr][i]; rowsum[rr] = s2; }
  __syncthreads();
  int mo = M0 + rr;
  if (mo < S) {
    float inv = 1.f / rowsum[rr];
    _Float16* prow = Pbuf + ((size_t)bh * S + mo) * PSTR;
    for (int n = gg; n < S; n += 8) {
      float p = sc[rr][n] * inv;
      scoresL[((size_t)bh * S + mo) * S + n] = p;
      prow[n] = (_Float16)p;
    }
    for (int n = S + gg; n < PSTR; n += 8) prow[n] = (_Float16)0.f;  // zero pad
  }
}

// ---------------- attention: O = P V ----------------
// grid (B*NH, 13), block 128 (4 waves, one 16-col HD subtile each).
// P rows padded to PSTR=224 with zeros -> aligned b128 A-frag loads, no k masking.
__global__ __launch_bounds__(128)
void k_attn_av(const _Float16* __restrict__ Pbuf, const _Float16* __restrict__ vbuf,
               _Float16* __restrict__ attn) {
  int bh = blockIdx.x;  int b = bh / NH;  int hh = bh % NH;
  int M0 = blockIdx.y * 16;
  int lane = threadIdx.x & 31;  int wv = threadIdx.x >> 5;
  int lm = lane & 15;
  int kb  = (lane >= 16) ? 8 : 0;
  int kb2 = (lane >= 16) ? 16 : 0;

  int m = M0 + lm;  int sm = (m < S) ? m : S - 1;
  const _Float16* prow = Pbuf + ((size_t)bh * S + sm) * PSTR;
  int n = wv * 16 + lm;  // HD column

  v8f_t acc = {};
  for (int kk = 0; kk < PSTR; kk += 32) {
    v16h_t a = make_a_frag(prow + kk, kb);      // zeros beyond k=196 nullify clamped V
    v16h_t bf;
#pragma unroll
    for (int i = 0; i < 16; ++i) {
      int kg = kk + kb2 + i;  if (kg >= S) kg = S - 1;   // clamp: stays in-bounds
      bf[i] = vbuf[((size_t)(b * S + kg) * D + hh * HD) + n];
    }
    acc = __builtin_amdgcn_wmma_f32_16x16x32_f16(false, a, false, bf,
                                                 (short)0, acc, false, false);
  }
  int mbase = M0 + ((lane >> 4) << 3);
#pragma unroll
  for (int r = 0; r < 8; ++r) {
    int mm = mbase + r;
    if (mm < S) attn[((size_t)(b * S + mm) * D + hh * HD) + n] = (_Float16)acc[r];
  }
}

// ---------------- host ----------------
extern "C" void kernel_launch(void* const* d_in, const int* in_sizes, int n_in,
                              void* d_out, int out_size, void* d_ws, size_t ws_size,
                              hipStream_t stream) {
  const float* x       = (const float*)d_in[0];
  const float* Wconv   = (const float*)d_in[1];
  const float* bconv   = (const float*)d_in[2];
  const float* cls_emb = (const float*)d_in[3];
  const float* pos_emb = (const float*)d_in[4];
  const float* Wq = (const float*)d_in[5];   const float* bq = (const float*)d_in[6];
  const float* Wk = (const float*)d_in[7];   const float* bk = (const float*)d_in[8];
  const float* Wv = (const float*)d_in[9];   const float* bv = (const float*)d_in[10];
  const float* Wo = (const float*)d_in[11];  const float* bo = (const float*)d_in[12];
  const float* ln1g = (const float*)d_in[13]; const float* ln1b = (const float*)d_in[14];
  const float* Wff1 = (const float*)d_in[15]; const float* bff1 = (const float*)d_in[16];
  const float* Wff2 = (const float*)d_in[17]; const float* bff2 = (const float*)d_in[18];
  const float* ln2g = (const float*)d_in[19]; const float* ln2b = (const float*)d_in[20];
  const float* lnfg = (const float*)d_in[21]; const float* lnfb = (const float*)d_in[22];
  const float* Whead = (const float*)d_in[23]; const float* bhead = (const float*)d_in[24];
  float* outF = (float*)d_out;

  // workspace carving (256B aligned)
  char* wsb = (char*)d_ws;
  size_t off = 0;
  auto alloc = [&](size_t bytes) -> void* {
    off = (off + 255) & ~(size_t)255;
    void* p = wsb + off;
    off += bytes;
    return p;
  };
  _Float16* im2A    = (_Float16*)alloc((size_t)BATCH * NP * D * 2);
  float*    patchT  = (float*)   alloc((size_t)BATCH * NP * D * 4);
  float*    hbuf    = (float*)   alloc((size_t)MROWS * D * 4);
  _Float16* n16     = (_Float16*)alloc((size_t)MROWS * D * 2);
  _Float16* q16     = (_Float16*)alloc((size_t)MROWS * D * 2);
  _Float16* k16     = (_Float16*)alloc((size_t)MROWS * D * 2);
  _Float16* v16     = (_Float16*)alloc((size_t)MROWS * D * 2);
  _Float16* attn16  = (_Float16*)alloc((size_t)MROWS * D * 2);
  _Float16* ff16    = (_Float16*)alloc((size_t)MROWS * DFF * 2);
  _Float16* P16     = (_Float16*)alloc((size_t)BATCH * NH * S * PSTR * 2);
  _Float16* cls16   = (_Float16*)alloc((size_t)BATCH * D * 2);
  _Float16* wconv16 = (_Float16*)alloc((size_t)D * D * 2);
  _Float16* whead16 = (_Float16*)alloc((size_t)NCLS * D * 2);
  _Float16* wq16    = (_Float16*)alloc((size_t)D * D * 2);
  _Float16* wk16    = (_Float16*)alloc((size_t)D * D * 2);
  _Float16* wv16    = (_Float16*)alloc((size_t)D * D * 2);
  _Float16* wo16    = (_Float16*)alloc((size_t)D * D * 2);
  _Float16* wff1h   = (_Float16*)alloc((size_t)DFF * D * 2);
  _Float16* wff2h   = (_Float16*)alloc((size_t)D * DFF * 2);

  auto cvt = [&](const float* s, _Float16* d, int n) {
    k_cvt_f32_f16<<<(n + 255) / 256, 256, 0, stream>>>(s, d, n);
  };
  auto gemm = [&](const _Float16* A, const _Float16* Wt, const float* bias,
                  void* out, int M, int N, int K, int mode) {
    dim3 g((N + 127) / 128, (M + 63) / 64);
    k_gemm_wmma<<<g, 256, 0, stream>>>(A, Wt, bias, (float*)out, (_Float16*)out, M, N, K, mode);
  };

  // --- patch embedding ---
  cvt(Wconv, wconv16, D * D);
  cvt(Whead, whead16, NCLS * D);
  {
    int n = BATCH * NP * D;
    k_im2col<<<(n + 255) / 256, 256, 0, stream>>>(x, im2A);
  }
  gemm(im2A, wconv16, bconv, patchT, BATCH * NP, D, D, /*f32*/1);
  {
    int n = MROWS * D;
    k_build_h<<<(n + 255) / 256, 256, 0, stream>>>(patchT, cls_emb, pos_emb, hbuf);
  }

  const size_t scoresPerLayer = (size_t)BATCH * NH * S * S;
  for (int l = 0; l < LAYERS; ++l) {
    cvt(Wq + (size_t)l * D * D, wq16, D * D);
    cvt(Wk + (size_t)l * D * D, wk16, D * D);
    cvt(Wv + (size_t)l * D * D, wv16, D * D);
    cvt(Wo + (size_t)l * D * D, wo16, D * D);
    cvt(Wff1 + (size_t)l * DFF * D, wff1h, DFF * D);
    cvt(Wff2 + (size_t)l * D * DFF, wff2h, D * DFF);

    k_layernorm_f16<<<MROWS, 256, 0, stream>>>(hbuf, ln1g + l * D, ln1b + l * D, n16, 1);
    gemm(n16, wq16, bq + l * D, q16, MROWS, D, D, /*f16*/0);
    gemm(n16, wk16, bk + l * D, k16, MROWS, D, D, 0);
    gemm(n16, wv16, bv + l * D, v16, MROWS, D, D, 0);

    float* scoresL = outF + (size_t)BATCH * NCLS + (size_t)l * scoresPerLayer;
    k_attn_scores<<<dim3(BATCH * NH, 13), 128, 0, stream>>>(q16, k16, scoresL, P16);
    k_attn_av<<<dim3(BATCH * NH, 13), 128, 0, stream>>>(P16, v16, attn16);

    gemm(attn16, wo16, bo + l * D, hbuf, MROWS, D, D, /*+= f32*/2);

    k_layernorm_f16<<<MROWS, 256, 0, stream>>>(hbuf, ln2g + l * D, ln2b + l * D, n16, 1);
    gemm(n16, wff1h, bff1 + l * DFF, ff16, MROWS, DFF, D, /*gelu f16*/3);
    gemm(ff16, wff2h, bff2 + l * D, hbuf, MROWS, D, DFF, 2);
  }

  // final LN on cls rows (row r -> h row r*S), then head GEMM -> logits
  k_layernorm_f16<<<BATCH, 256, 0, stream>>>(hbuf, lnfg, lnfb, cls16, S);
  gemm(cls16, whead16, bhead, outF, BATCH, NCLS, D, /*f32*/1);
}
